// Block_14663018348733
// MI455X (gfx1250) — matrix operand order repaired and checked by reading, not compiled, not run on hardware
//
#include <hip/hip_runtime.h>
#include <hip/hip_bf16.h>

// ---- problem dims (compile-time) ----
constexpr int BB = 2, TT = 2048, CC = 768, HH = 12, DD = 64;
constexpr int MROWS = BB * TT;          // 4096
constexpr float EPS = 1e-5f;

typedef float  v8f   __attribute__((ext_vector_type(8)));
typedef __bf16 v16bf __attribute__((ext_vector_type(16)));
typedef __bf16 v8bf  __attribute__((ext_vector_type(8)));
typedef __bf16 v4bf  __attribute__((ext_vector_type(4)));
typedef unsigned int v4u __attribute__((ext_vector_type(4)));
typedef int v4i __attribute__((ext_vector_type(4)));
typedef int v8i __attribute__((ext_vector_type(8)));

#if __has_builtin(__builtin_amdgcn_tensor_load_to_lds)
#define HAVE_TDM 1
#else
#define HAVE_TDM 0
#endif

__device__ __forceinline__ __bf16 f2bf(float x) {
    unsigned u = __builtin_bit_cast(unsigned, x);
    u += 0x7FFFu + ((u >> 16) & 1u);               // round-to-nearest-even
    return __builtin_bit_cast(__bf16, (unsigned short)(u >> 16));
}

__device__ __forceinline__ v16bf cat8(v8bf lo, v8bf hi) {
    return __builtin_shufflevector(lo, hi, 0,1,2,3,4,5,6,7,8,9,10,11,12,13,14,15);
}

#if HAVE_TDM
// 2D TDM load: tile_rows x tile_k bf16 elements, global row stride = stride
// elems, packed contiguously into LDS (row-major [tile_rows][tile_k]).
// D# layout per CDNA5 ISA ch.8 (group0: flags/lds/global addr, group1: dims).
__device__ __forceinline__ void tdm_load_2d(const void* gsrc, void* ldst,
                                            unsigned tile_k, unsigned tile_rows,
                                            unsigned stride_elems)
{
    unsigned long long ga = (unsigned long long)gsrc;
    v4u g0;
    g0.x = 1u;                                           // count=1 (valid), user mode
    g0.y = (unsigned)(unsigned long long)ldst;           // lds_addr (bytes)
    g0.z = (unsigned)ga;                                 // global_addr[31:0]
    g0.w = (unsigned)((ga >> 32) & 0x01FFFFFFu) | (2u << 30); // addr[56:32] | type=2
    v8i g1;
    g1[0] = (int)(1u << 16);                             // data_size=2B, mask=0
    g1[1] = (int)((stride_elems & 0xFFFFu) << 16);       // tensor_dim0 lo16
    g1[2] = (int)((stride_elems >> 16) & 0xFFFFu);       // tensor_dim0 hi16, dim1 lo16=0
    g1[3] = (int)(0x10u | (tile_k << 16));               // tensor_dim1 hi16 (1M rows), tile_dim0
    g1[4] = (int)(tile_rows & 0xFFFFu);                  // tile_dim1, tile_dim2=0
    g1[5] = (int)stride_elems;                           // tensor_dim0_stride lo32
    g1[6] = 0;                                           // stride hi, dim1_stride lo
    g1[7] = 0;
    v4i z4 = {0, 0, 0, 0};
#if defined(__clang_major__) && (__clang_major__ >= 23)
    v8i z8 = {0, 0, 0, 0, 0, 0, 0, 0};
    __builtin_amdgcn_tensor_load_to_lds(g0, g1, z4, z4, z8, 0);
#else
    __builtin_amdgcn_tensor_load_to_lds(g0, g1, z4, z4, 0);
#endif
}
#endif

// fallback staging helper: copy tile_rows x WELEM bf16 tile into LDS
template<int WELEM>
__device__ __forceinline__ void stage_tile_bf16(const __bf16* __restrict__ g,
                                                size_t gstride, __bf16* l,
                                                int rows, int tid, int nthr)
{
    const int chunks = WELEM / 8;
    for (int i = tid; i < rows * chunks; i += nthr) {
        int r = i / chunks, c = (i % chunks) * 8;
        *reinterpret_cast<v8bf*>(l + r * WELEM + c) =
            *reinterpret_cast<const v8bf*>(g + (size_t)r * gstride + c);
    }
}

// =====================================================================
// Elementwise convert f32 -> bf16 (4 per thread)
// =====================================================================
__global__ __launch_bounds__(256) void cvt_f32_bf16(
    const float* __restrict__ x, __bf16* __restrict__ y)
{
    int i = (blockIdx.x * 256 + threadIdx.x) * 4;
    float4 f = *reinterpret_cast<const float4*>(x + i);
    y[i + 0] = f2bf(f.x); y[i + 1] = f2bf(f.y);
    y[i + 2] = f2bf(f.z); y[i + 3] = f2bf(f.w);
}

// Transpose + convert weights: Wt[n][k] = bf16(W[k][n])
__global__ __launch_bounds__(256) void transpose_f32_bf16(
    const float* __restrict__ W, __bf16* __restrict__ Wt, int K, int N)
{
    int id = blockIdx.x * 256 + threadIdx.x;   // one per (k, n/4)
    int nq = N >> 2;
    int k = id / nq, n = (id % nq) << 2;
    float4 f = *reinterpret_cast<const float4*>(W + (size_t)k * N + n);
    Wt[(size_t)(n + 0) * K + k] = f2bf(f.x);
    Wt[(size_t)(n + 1) * K + k] = f2bf(f.y);
    Wt[(size_t)(n + 2) * K + k] = f2bf(f.z);
    Wt[(size_t)(n + 3) * K + k] = f2bf(f.w);
}

// =====================================================================
// bf16 WMMA GEMM, TDM-staged, double buffered.
//   out = epi(A[M,K] @ Bt[N,K]^T + bias [+ R])
// Block tile 128x256, 8 waves (2x4), wave tile 64x64 = 4x4 WMMA accums.
// EPI: 0 = bias -> bf16 out; 1 = bias + residual -> f32 out;
//      2 = bias + exact GELU -> bf16 out
// =====================================================================
template<int EPI>
__global__ __launch_bounds__(256) void gemm_bf16_wmma(
    const __bf16* __restrict__ A, const __bf16* __restrict__ Bt,
    const float* __restrict__ bias, const float* __restrict__ R,
    float* __restrict__ outf, __bf16* __restrict__ outb,
    int M, int N, int K)
{
    __shared__ __bf16 As[2][128][32];   // 16 KB
    __shared__ __bf16 Bs[2][256][32];   // 32 KB

    const int tid  = threadIdx.x;
    const int wave = tid >> 5, lane = tid & 31;
    const int hl   = lane >> 4, l15 = lane & 15;
    const int wrow = wave >> 2, wcol = wave & 3;          // 2x4 wave grid
    const int mblk = blockIdx.y * 128, nblk = blockIdx.x * 256;
    const int nk = K / 32;

    v8f acc[4][4] = {};

#if HAVE_TDM
    if (wave == 0) {
        tdm_load_2d(A  + (size_t)mblk * K, &As[0][0][0], 32, 128, (unsigned)K);
        tdm_load_2d(Bt + (size_t)nblk * K, &Bs[0][0][0], 32, 256, (unsigned)K);
    }
#endif
    for (int t = 0; t < nk; ++t) {
#if HAVE_TDM
        const int cur = t & 1;
        if (wave == 0) {
            if (t + 1 < nk) {   // prefetch next tile into other buffer via TDM
                int k0n = (t + 1) * 32;
                tdm_load_2d(A  + (size_t)mblk * K + k0n, &As[1 - cur][0][0], 32, 128, (unsigned)K);
                tdm_load_2d(Bt + (size_t)nblk * K + k0n, &Bs[1 - cur][0][0], 32, 256, (unsigned)K);
                __builtin_amdgcn_s_wait_tensorcnt(2);   // current tile landed
            } else {
                __builtin_amdgcn_s_wait_tensorcnt(0);
            }
        }
        __syncthreads();
#else
        const int cur = 0;
        __syncthreads();
        stage_tile_bf16<32>(A  + (size_t)mblk * K + t * 32, K, &As[0][0][0], 128, tid, 256);
        stage_tile_bf16<32>(Bt + (size_t)nblk * K + t * 32, K, &Bs[0][0][0], 256, tid, 256);
        __syncthreads();
#endif
        v16bf af[4], bfrag[4];
        #pragma unroll
        for (int i = 0; i < 4; ++i) {   // A frag: M=l15, 16 contiguous K per lane
            int row = wrow * 64 + i * 16 + l15;
            v8bf lo = *reinterpret_cast<const v8bf*>(&As[cur][row][hl * 16]);
            v8bf hi = *reinterpret_cast<const v8bf*>(&As[cur][row][hl * 16 + 8]);
            af[i] = cat8(lo, hi);
        }
        #pragma unroll
        for (int j = 0; j < 4; ++j) {   // B frag: N=l15, 16 contiguous K per lane
            int col = wcol * 64 + j * 16 + l15;
            bfrag[j] = *reinterpret_cast<const v16bf*>(&Bs[cur][col][hl * 16]);
        }
        #pragma unroll
        for (int i = 0; i < 4; ++i)
            #pragma unroll
            for (int j = 0; j < 4; ++j)
                acc[i][j] = __builtin_amdgcn_wmma_f32_16x16x32_bf16(
                    false, af[i], false, bfrag[j], (short)0, acc[i][j], false, false);
        __syncthreads();
    }

    // epilogue: C layout -> M = r + 8*hl, N = l15
    #pragma unroll
    for (int j = 0; j < 4; ++j) {
        int gcol = nblk + wcol * 64 + j * 16 + l15;
        float bj = bias[gcol];
        #pragma unroll
        for (int i = 0; i < 4; ++i) {
            #pragma unroll
            for (int r = 0; r < 8; ++r) {
                int grow = mblk + wrow * 64 + i * 16 + r + 8 * hl;
                float v = acc[i][j][r] + bj;
                if (EPI == 1) {
                    v += R[(size_t)grow * N + gcol];
                    outf[(size_t)grow * N + gcol] = v;
                } else {
                    if (EPI == 2) v = 0.5f * v * (1.0f + erff(v * 0.70710678118f));
                    outb[(size_t)grow * N + gcol] = f2bf(v);
                }
            }
        }
    }
}

// =====================================================================
// Flash attention (causal), bf16 qkv in, bf16 out.
// Block = 64 query rows of one (b,h); 4 waves x 16 query rows.
// 1/sqrt(D) folded into score scaling post-WMMA.
// =====================================================================
__global__ __launch_bounds__(128) void attn_flash(
    const __bf16* __restrict__ qkv, __bf16* __restrict__ out)
{
    __shared__ __bf16 Qs[64][64];       // [q][d]
    __shared__ __bf16 Ks[64][64];       // [key][d]  (== Bt layout for QK^T)
    __shared__ __bf16 Vt[64][64];       // [d][key]  (== Bt layout for P@V)
    __shared__ __bf16 Ps[4][16][64];    // per-wave P re-layout buffer

    const int tid  = threadIdx.x;
    const int wave = tid >> 5, lane = tid & 31;
    const int hl   = lane >> 4, l15 = lane & 15;
    const int qt = blockIdx.x, h = blockIdx.y, b = blockIdx.z;
    const size_t rs   = 3 * CC;                      // qkv row stride
    const size_t base = (size_t)b * TT * rs;
    const int qcol = h * DD, kcol = CC + h * DD, vcol = 2 * CC + h * DD;

    // load Q tile
#if HAVE_TDM
    if (wave == 0) {
        tdm_load_2d(qkv + base + (size_t)(qt * 64) * rs + qcol, &Qs[0][0], 64, 64, (unsigned)rs);
        __builtin_amdgcn_s_wait_tensorcnt(0);
    }
#else
    stage_tile_bf16<64>(qkv + base + (size_t)(qt * 64) * rs + qcol, rs, &Qs[0][0], 64, tid, 128);
#endif
    __syncthreads();

    v16bf qa[2];
    #pragma unroll
    for (int s = 0; s < 2; ++s) {
        int row = wave * 16 + l15;
        v8bf lo = *reinterpret_cast<const v8bf*>(&Qs[row][s * 32 + hl * 16]);
        v8bf hi = *reinterpret_cast<const v8bf*>(&Qs[row][s * 32 + hl * 16 + 8]);
        qa[s] = cat8(lo, hi);
    }

    v8f oacc[4] = {};
    float mrow[8], lrow[8];
    #pragma unroll
    for (int r = 0; r < 8; ++r) { mrow[r] = -3.0e38f; lrow[r] = 0.0f; }

    for (int kt = 0; kt <= qt; ++kt) {
        __syncthreads();
#if HAVE_TDM
        if (wave == 0)
            tdm_load_2d(qkv + base + (size_t)(kt * 64) * rs + kcol, &Ks[0][0], 64, 64, (unsigned)rs);
#else
        stage_tile_bf16<64>(qkv + base + (size_t)(kt * 64) * rs + kcol, rs, &Ks[0][0], 64, tid, 128);
#endif
        {   // stage V transposed (manual: TDM cannot transpose)
            int r = tid >> 4, c = (tid & 15) * 4;
            #pragma unroll
            for (int p = 0; p < 8; ++p) {
                int row = r + p * 8;
                v4bf g = *reinterpret_cast<const v4bf*>(
                    qkv + base + (size_t)(kt * 64 + row) * rs + vcol + c);
                Vt[c + 0][row] = g[0]; Vt[c + 1][row] = g[1];
                Vt[c + 2][row] = g[2]; Vt[c + 3][row] = g[3];
            }
        }
#if HAVE_TDM
        if (wave == 0) __builtin_amdgcn_s_wait_tensorcnt(0);
#endif
        __syncthreads();

        // S = Q K^T (unscaled), 16 q-rows per wave x 64 keys
        v8f sacc[4] = {};
        #pragma unroll
        for (int s = 0; s < 2; ++s)
            #pragma unroll
            for (int j = 0; j < 4; ++j) {
                v16bf kb = *reinterpret_cast<const v16bf*>(
                    &Ks[j * 16 + l15][s * 32 + hl * 16]);
                sacc[j] = __builtin_amdgcn_wmma_f32_16x16x32_bf16(
                    false, qa[s], false, kb, (short)0, sacc[j], false, false);
            }
        #pragma unroll
        for (int j = 0; j < 4; ++j)
            #pragma unroll
            for (int r = 0; r < 8; ++r)
                sacc[j][r] *= 0.125f;           // 1/sqrt(64)

        if (kt == qt) {  // causal mask on diagonal tile
            #pragma unroll
            for (int j = 0; j < 4; ++j) {
                int n = j * 16 + l15;
                #pragma unroll
                for (int r = 0; r < 8; ++r) {
                    int m = wave * 16 + r + 8 * hl;
                    if (n > m) sacc[j][r] = -3.0e38f;
                }
            }
        }

        // online softmax (row = fixed (r, half); reduce across 16 lanes of half)
        #pragma unroll
        for (int r = 0; r < 8; ++r) {
            float mx = fmaxf(fmaxf(sacc[0][r], sacc[1][r]),
                             fmaxf(sacc[2][r], sacc[3][r]));
            #pragma unroll
            for (int off = 1; off < 16; off <<= 1)
                mx = fmaxf(mx, __shfl_xor(mx, off, 32));
            float mnew  = fmaxf(mrow[r], mx);
            float alpha = expf(mrow[r] - mnew);
            mrow[r] = mnew;
            float ps = 0.0f;
            #pragma unroll
            for (int j = 0; j < 4; ++j) {
                float p = expf(sacc[j][r] - mnew);
                ps += p;
                Ps[wave][r + 8 * hl][j * 16 + l15] = f2bf(p);
            }
            #pragma unroll
            for (int off = 1; off < 16; off <<= 1)
                ps += __shfl_xor(ps, off, 32);
            lrow[r] = lrow[r] * alpha + ps;
            #pragma unroll
            for (int j = 0; j < 4; ++j) oacc[j][r] *= alpha;
        }

        // O += P @ V   (wave-private Ps; same-wave LDS RAW handled by dscnt)
        #pragma unroll
        for (int s = 0; s < 2; ++s) {
            v8bf lo = *reinterpret_cast<const v8bf*>(&Ps[wave][l15][s * 32 + hl * 16]);
            v8bf hi = *reinterpret_cast<const v8bf*>(&Ps[wave][l15][s * 32 + hl * 16 + 8])
            v16bf pa = cat8(lo, hi);
            #pragma unroll
            for (int j = 0; j < 4; ++j) {
                v16bf vb = *reinterpret_cast<const v16bf*>(
                    &Vt[j * 16 + l15][s * 32 + hl * 16]);
                oacc[j] = __builtin_amdgcn_wmma_f32_16x16x32_bf16(
                    false, pa, false, vb, (short)0, oacc[j], false, false);
            }
        }
    }

    // out[b, q, h*64 + d] (bf16, feeds Wo GEMM)
    #pragma unroll
    for (int j = 0; j < 4; ++j)
        #pragma unroll
        for (int r = 0; r < 8; ++r) {
            int m = wave * 16 + r + 8 * hl;
            int d = j * 16 + l15;
            size_t row = (size_t)b * TT + qt * 64 + m;
            out[row * CC + h * DD + d] = f2bf(oacc[j][r] / lrow[r]);
        }
}

// =====================================================================
// Row-wise LayerNorm, 1 block per row (C = 768 = 3 * 256).
// Optionally also emits a bf16 copy (for the next GEMM's A operand).
// =====================================================================
__global__ __launch_bounds__(256) void layernorm_k(
    const float* __restrict__ x, const float* __restrict__ g,
    const float* __restrict__ be, float* __restrict__ outf,
    __bf16* __restrict__ outb)
{
    __shared__ float red[8];
    const int tid = threadIdx.x, wave = tid >> 5, lane = tid & 31;
    const float* xr = x + (size_t)blockIdx.x * CC;
    float v0 = xr[tid], v1 = xr[tid + 256], v2 = xr[tid + 512];

    float s = v0 + v1 + v2;
    #pragma unroll
    for (int o = 1; o < 32; o <<= 1) s += __shfl_xor(s, o, 32);
    if (lane == 0) red[wave] = s;
    __syncthreads();
    float tot = 0.0f;
    #pragma unroll
    for (int i = 0; i < 8; ++i) tot += red[i];
    float mu = tot * (1.0f / (float)CC);

    float d0 = v0 - mu, d1 = v1 - mu, d2 = v2 - mu;
    float q = d0 * d0 + d1 * d1 + d2 * d2;
    #pragma unroll
    for (int o = 1; o < 32; o <<= 1) q += __shfl_xor(q, o, 32);
    __syncthreads();
    if (lane == 0) red[wave] = q;
    __syncthreads();
    float tot2 = 0.0f;
    #pragma unroll
    for (int i = 0; i < 8; ++i) tot2 += red[i];
    float rstd = rsqrtf(tot2 * (1.0f / (float)CC) + EPS);

    float y0 = d0 * rstd * g[tid]       + be[tid];
    float y1 = d1 * rstd * g[tid + 256] + be[tid + 256];
    float y2 = d2 * rstd * g[tid + 512] + be[tid + 512];
    float* o = outf + (size_t)blockIdx.x * CC;
    o[tid] = y0; o[tid + 256] = y1; o[tid + 512] = y2;
    if (outb) {
        __bf16* ob = outb + (size_t)blockIdx.x * CC;
        ob[tid] = f2bf(y0); ob[tid + 256] = f2bf(y1); ob[tid + 512] = f2bf(y2);
    }
}

// =====================================================================
extern "C" void kernel_launch(void* const* d_in, const int* in_sizes, int n_in,
                              void* d_out, int out_size, void* d_ws, size_t ws_size,
                              hipStream_t stream) {
    (void)in_sizes; (void)n_in; (void)out_size; (void)ws_size;
    const float* x    = (const float*)d_in[0];
    const float* Wqkv = (const float*)d_in[1];
    const float* bqkv = (const float*)d_in[2];
    const float* Wo   = (const float*)d_in[3];
    const float* bo   = (const float*)d_in[4];
    const float* g1   = (const float*)d_in[5];
    const float* b1   = (const float*)d_in[6];
    const float* Wf   = (const float*)d_in[7];
    const float* bfv  = (const float*)d_in[8];
    const float* Wp   = (const float*)d_in[9];
    const float* bp   = (const float*)d_in[10];
    const float* g2   = (const float*)d_in[11];
    const float* b2   = (const float*)d_in[12];
    float* out = (float*)d_out;

    // ---- workspace carve (f32 region first, then bf16 region) ----
    char* p = (char*)d_ws;
    auto takef = [&](size_t n) { float* q = (float*)p; p += n * sizeof(float); return q; };
    float* h1  = takef((size_t)MROWS * CC);
    float* ln1 = takef((size_t)MROWS * CC);
    float* h2  = takef((size_t)MROWS * CC);
    auto takeb = [&](size_t n) { __bf16* q = (__bf16*)p; p += n * sizeof(__bf16); return q; };
    __bf16* xbf    = takeb((size_t)MROWS * CC);
    __bf16* Wqkvt  = takeb((size_t)CC * 3 * CC);
    __bf16* Wot    = takeb((size_t)CC * CC);
    __bf16* Wft    = takeb((size_t)CC * 4 * CC);
    __bf16* Wpt    = takeb((size_t)4 * CC * CC);
    __bf16* qkvb   = takeb((size_t)MROWS * 3 * CC);
    __bf16* attnb  = takeb((size_t)MROWS * CC);
    __bf16* ln1b   = takeb((size_t)MROWS * CC);
    __bf16* ffnb   = takeb((size_t)MROWS * 4 * CC);

    dim3 blk(256);
    // one-time precision/layout conversion
    cvt_f32_bf16<<<(MROWS * CC) / 1024, blk, 0, stream>>>(x, xbf);
    transpose_f32_bf16<<<(CC * 3 * CC) / 1024, blk, 0, stream>>>(Wqkv, Wqkvt, CC, 3 * CC);
    transpose_f32_bf16<<<(CC * CC) / 1024, blk, 0, stream>>>(Wo, Wot, CC, CC);
    transpose_f32_bf16<<<(CC * 4 * CC) / 1024, blk, 0, stream>>>(Wf, Wft, CC, 4 * CC);
    transpose_f32_bf16<<<(4 * CC * CC) / 1024, blk, 0, stream>>>(Wp, Wpt, 4 * CC, CC);

    // qkv = x @ Wqkv + bqkv                (bf16 out)
    gemm_bf16_wmma<0><<<dim3((3 * CC) / 256, MROWS / 128), blk, 0, stream>>>(
        xbf, Wqkvt, bqkv, nullptr, nullptr, qkvb, MROWS, 3 * CC, CC);
    // attn = causal_softmax(q k^T / 8) v   (bf16 out)
    attn_flash<<<dim3(TT / 64, HH, BB), dim3(128), 0, stream>>>(qkvb, attnb);
    // h1 = x + attn @ Wo + bo              (f32 out)
    gemm_bf16_wmma<1><<<dim3(CC / 256, MROWS / 128), blk, 0, stream>>>(
        attnb, Wot, bo, x, h1, nullptr, MROWS, CC, CC);
    layernorm_k<<<dim3(MROWS), blk, 0, stream>>>(h1, g1, b1, ln1, ln1b);
    // ffn = gelu(ln1 @ Wf + bf)            (bf16 out)
    gemm_bf16_wmma<2><<<dim3((4 * CC) / 256, MROWS / 128), blk, 0, stream>>>(
        ln1b, Wft, bfv, nullptr, nullptr, ffnb, MROWS, 4 * CC, CC);
    // h2 = ln1 + ffn @ Wp + bp             (f32 out)
    gemm_bf16_wmma<1><<<dim3(CC / 256, MROWS / 128), blk, 0, stream>>>(
        ffnb, Wpt, bp, ln1, h2, nullptr, MROWS, CC, 4 * CC);
    layernorm_k<<<dim3(MROWS), blk, 0, stream>>>(h2, g2, b2, out, nullptr);
}
